// LSTM_51410758533695
// MI455X (gfx1250) — matrix operand order repaired and checked by reading, not compiled, not run on hardware
//
#include <hip/hip_runtime.h>
#include <math.h>
#include <stdint.h>

// ---------------------------------------------------------------------------
// LSTM for MI455X (gfx1250): bf16 WMMA GEMMs + TDM async tile staging +
// fused recurrent step kernel.
//   inputs (fp32): x_t,h0,c0, We,be, Wf,bf, Wi,bi, Wg,bg, Wo,bo, Rf,Ri,Rg,Ro
//   d_out (fp32): hidden_seq[B,T,H] ++ h_last[B,H] ++ c_last[B,H]
// ---------------------------------------------------------------------------

typedef __bf16 bf16_t;
typedef bf16_t v16bf __attribute__((ext_vector_type(16)));
typedef bf16_t v8bf  __attribute__((ext_vector_type(8)));
typedef float  v8f   __attribute__((ext_vector_type(8)));
typedef unsigned int u32;
typedef u32  u32x4 __attribute__((ext_vector_type(4)));
typedef int  i32x4 __attribute__((ext_vector_type(4)));
typedef int  i32x8 __attribute__((ext_vector_type(8)));

namespace {
constexpr int kB   = 128;
constexpr int kT   = 512;
constexpr int kDIM = 256;
constexpr int kE   = 512;
constexpr int kH   = 512;
constexpr int kNH  = 4 * kH;       // 2048 (gate order: i, f, g, o)
constexpr int kM   = kB * kT;      // 65536 rows for the batched GEMMs
}

#define DEVFN static __device__ __forceinline__

DEVFN v8f wmma_bf16(v16bf a, v16bf b, v8f c) {
  // (neg_a, A, neg_b, B, c_mod, C, reuse_a, reuse_b)
  return __builtin_amdgcn_wmma_f32_16x16x32_bf16(false, a, false, b, (short)0, c,
                                                 false, false);
}

// A-matrix fragment (16x32 bf16, row-major source, row stride `stride`):
// lane m (0-15) / m+16 -> 16 contiguous K values (two 16 B chunks), the lane
// half shifting K by +16 (matches B fragment order).
DEVFN v16bf load_a_frag(const bf16_t* __restrict__ base, int row, int stride,
                        int k0, int laneHi) {
  union { v16bf v; struct { v8bf lo, hi; } s; } u;
  const bf16_t* p0 = base + (size_t)row * stride + k0 + 16 * laneHi;
  u.s.lo = *reinterpret_cast<const v8bf*>(p0);
  u.s.hi = *reinterpret_cast<const v8bf*>(p0 + 8);
  return u.v;
}

// B-matrix fragment (32x16 bf16): weights stored [N,K] row-major; a lane
// holds 16 contiguous K values of output column n = lane%16, the lane half
// selecting the K half (contiguous order, same as the A fragment).
DEVFN v16bf load_b_frag(const bf16_t* __restrict__ wrow, int k0, int laneHi) {
  return *reinterpret_cast<const v16bf*>(wrow + k0 + 16 * laneHi);
}

DEVFN float sigmoid_f(float x) { return 1.0f / (1.0f + __expf(-x)); }

// ---------------------------------------------------------------------------
// Tensor Data Mover: 1-D tile load global -> LDS.
// D# built per ISA 8.3/8.4: data_size=8B, tile_dim0 = nbytes/8 (1-D tile),
// tensor_dim0 = stride0 = nelem, tensor_dim1 = 1. Per-wave op: call from one
// wave, then s_wait_tensorcnt 0 + workgroup barrier before consuming.
DEVFN void tdm_load_1d(u32 lds_off, const void* src, u32 nbytes) {
  const unsigned long long ga = (unsigned long long)(uintptr_t)src;
  const u32 nelem = nbytes >> 3;                       // 8-byte elements
  u32x4 g0;
  g0.x = 1u;                                           // count=1 (valid D#)
  g0.y = lds_off;                                      // LDS byte address
  g0.z = (u32)(ga & 0xFFFFFFFFu);                      // global_addr[31:0]
  g0.w = (u32)((ga >> 32) & 0x01FFFFFFu) | (2u << 30); // addr[56:32] | type=2

  i32x8 g1;
  g1[0] = (int)(3u << 16);                             // data_size = 8 bytes
  g1[1] = (int)((nelem & 0xFFFFu) << 16);              // tensor_dim0[15:0]
  g1[2] = (int)(((nelem >> 16) & 0xFFFFu) | (1u << 16)); // dim0 hi | tensor_dim1=1
  g1[3] = (int)((nelem & 0xFFFFu) << 16);              // tile_dim0 (1-D tile)
  g1[4] = 0;                                           // tile_dim1/2 unused
  g1[5] = (int)nelem;                                  // tensor_dim0_stride lo
  g1[6] = 0;
  g1[7] = 0;

  i32x4 z4 = {0, 0, 0, 0};
#if __has_include(<hip/amd_detail/amd_gfx1250_TDM.h>)
  i32x8 z8 = {0, 0, 0, 0, 0, 0, 0, 0};
  __builtin_amdgcn_tensor_load_to_lds(g0, g1, z4, z4, z8, 0);
#else
  __builtin_amdgcn_tensor_load_to_lds(g0, g1, z4, z4, 0);
#endif
}

// Low 32 bits of a flat shared-memory address are the LDS byte offset
// (aperture check uses addr[63:32] only).
DEVFN u32 lds_addr_of(const void* p) { return (u32)(uintptr_t)p; }

// ---------------------------------------------------------------------------
// Elementwise helpers
// ---------------------------------------------------------------------------
__global__ void f32_to_bf16_kernel(const float* __restrict__ src,
                                   bf16_t* __restrict__ dst, int n) {
  int i = blockIdx.x * blockDim.x + threadIdx.x;
  if (i < n) dst[i] = (bf16_t)src[i];
}

__global__ void copy_f32_kernel(const float* __restrict__ src,
                                float* __restrict__ dst, int n) {
  int i = blockIdx.x * blockDim.x + threadIdx.x;
  if (i < n) dst[i] = src[i];
}

// ---------------------------------------------------------------------------
// Batched GEMM: Out[M,N] = A[M,K] * Bw[N,K]^T + bias[N]
// One block = one 16-row M tile (TDM-staged in LDS); 8 waves, each wave
// register-blocks 4 N-tiles (16x64 output) so one A fragment feeds 4 WMMAs.
// ---------------------------------------------------------------------------
template <bool OUT_BF16>
__global__ void gemm_bf16_kernel(const bf16_t* __restrict__ A,
                                 const bf16_t* __restrict__ Bw,
                                 const float* __restrict__ bias,
                                 void* __restrict__ Out,
                                 int N, int K) {
  extern __shared__ bf16_t sA[];            // [16 * K]
  const int mtile  = blockIdx.x;
  const int tid    = threadIdx.x;           // 256 threads = 8 waves
  const int wave   = tid >> 5;
  const int lane   = tid & 31;
  const int laneHi = lane >> 4;
  const int lane16 = lane & 15;

  if (wave == 0) {                           // async DMA of the 16xK A tile
    tdm_load_1d(lds_addr_of(sA), A + (size_t)mtile * 16 * K,
                (u32)(16 * K * sizeof(bf16_t)));
    __builtin_amdgcn_s_wait_tensorcnt(0);
  }
  __syncthreads();

  for (int n0 = wave * 64; n0 < N; n0 += 8 * 64) {
    v8f acc[4];
#pragma unroll
    for (int j = 0; j < 4; ++j)
      acc[j] = (v8f){0.f, 0.f, 0.f, 0.f, 0.f, 0.f, 0.f, 0.f};
    const bf16_t* wrow = Bw + (size_t)(n0 + lane16) * K;
    for (int k0 = 0; k0 < K; k0 += 32) {
      __builtin_prefetch(wrow + k0 + 256, 0, 1);       // global_prefetch_b8
      const v16bf af = load_a_frag(sA, lane16, K, k0, laneHi);
#pragma unroll
      for (int j = 0; j < 4; ++j) {
        const v16bf bfrag = load_b_frag(wrow + (size_t)j * 16 * K, k0, laneHi);
        acc[j] = wmma_bf16(af, bfrag, acc[j]);
      }
    }
#pragma unroll
    for (int j = 0; j < 4; ++j) {
      const int n = n0 + j * 16 + lane16;
      const float bv = bias ? bias[n] : 0.f;
#pragma unroll
      for (int r = 0; r < 8; ++r) {
        const int m = r + 8 * laneHi;       // C/D layout: vgpr r -> row
        const size_t off = (size_t)(mtile * 16 + m) * N + n;
        const float val = acc[j][r] + bv;
        if (OUT_BF16) ((bf16_t*)Out)[off] = (bf16_t)val;
        else          ((float*)Out)[off]  = val;
      }
    }
  }
}

// ---------------------------------------------------------------------------
// One recurrence step:
//   pre[g] = xs[:,t,g,:] + h_prev @ R_g^T       (WMMA, gate per wave)
//   i,f,o = sigmoid; g = tanh; c = g*i + f*c;  h = o*tanh(c)
// grid = (B/16, H/32), 128 threads (4 waves; wave == gate, 2 h-tiles/wave).
// ---------------------------------------------------------------------------
__global__ void lstm_step_kernel(const bf16_t* __restrict__ hprev,  // [B,H] bf16
                                 bf16_t* __restrict__ hnext,        // [B,H] bf16
                                 const bf16_t* __restrict__ Rcat,   // [4H,H] bf16
                                 const float* __restrict__ xs,      // [B*T,4H] f32
                                 float* __restrict__ cstate,        // [B,H] f32
                                 float* __restrict__ hseq,          // [B,T,H] f32
                                 int t) {
  __shared__ bf16_t sH[16 * kH];      // 16 KB: h tile rows (TDM-staged)
  __shared__ float  sG[4][16][32];    // gate preacts for exchange

  const int mtile  = blockIdx.x;      // batch tile (0..7)
  const int hbase  = blockIdx.y * 32; // hidden column base (0..480 step 32)
  const int tid    = threadIdx.x;     // 0..127
  const int gate   = tid >> 5;        // wave id == gate (i,f,g,o)
  const int lane   = tid & 31;
  const int laneHi = lane >> 4;
  const int lane16 = lane & 15;

  if (gate == 0) {                    // 16 contiguous rows of h_prev -> LDS
    tdm_load_1d(lds_addr_of(sH), hprev + (size_t)mtile * 16 * kH,
                (u32)(16 * kH * sizeof(bf16_t)));
    __builtin_amdgcn_s_wait_tensorcnt(0);
  }
  __syncthreads();

  const bf16_t* wrow =
      Rcat + (size_t)(gate * kH + hbase + lane16) * kH;

  v8f acc[2];
#pragma unroll
  for (int j = 0; j < 2; ++j)
    acc[j] = (v8f){0.f, 0.f, 0.f, 0.f, 0.f, 0.f, 0.f, 0.f};
  for (int k0 = 0; k0 < kH; k0 += 32) {
    __builtin_prefetch(wrow + k0 + 256, 0, 1);
    const v16bf af = load_a_frag(sH, lane16, kH, k0, laneHi);
#pragma unroll
    for (int j = 0; j < 2; ++j) {
      const v16bf bfrag = load_b_frag(wrow + (size_t)j * 16 * kH, k0, laneHi);
      acc[j] = wmma_bf16(af, bfrag, acc[j]);
    }
  }

  // Add precomputed input projection and park preacts in LDS.
#pragma unroll
  for (int j = 0; j < 2; ++j) {
    const int n = hbase + j * 16 + lane16;
#pragma unroll
    for (int r = 0; r < 8; ++r) {
      const int m = r + 8 * laneHi;
      const int b = mtile * 16 + m;
      const float pre =
          acc[j][r] + xs[((size_t)b * kT + t) * kNH + (gate * kH + n)];
      sG[gate][m][j * 16 + lane16] = pre;
    }
  }
  __syncthreads();

  // Fused gate math: 128 threads handle 16x32 = 512 cells.
#pragma unroll
  for (int e = 0; e < 4; ++e) {
    const int idx = tid + e * 128;
    const int m = idx >> 5, nn = idx & 31;
    const int b = mtile * 16 + m;
    const int hcol = hbase + nn;
    const float ig = sigmoid_f(sG[0][m][nn]);
    const float fg = sigmoid_f(sG[1][m][nn]);
    const float gg = tanhf(sG[2][m][nn]);
    const float og = sigmoid_f(sG[3][m][nn]);
    const size_t ch = (size_t)b * kH + hcol;
    const float c = gg * ig + fg * cstate[ch];
    cstate[ch] = c;
    const float h = og * tanhf(c);
    hseq[((size_t)b * kT + t) * kH + hcol] = h;
    hnext[ch] = (bf16_t)h;
  }
}

__global__ void finalize_kernel(const float* __restrict__ hseq,
                                const float* __restrict__ cst,
                                float* __restrict__ hlast,
                                float* __restrict__ clast) {
  int i = blockIdx.x * blockDim.x + threadIdx.x;
  if (i < kB * kH) {
    const int b = i / kH, h = i % kH;
    hlast[i] = hseq[((size_t)b * kT + (kT - 1)) * kH + h];
    clast[i] = cst[i];
  }
}

// ---------------------------------------------------------------------------
extern "C" void kernel_launch(void* const* d_in, const int* in_sizes, int n_in,
                              void* d_out, int out_size, void* d_ws,
                              size_t ws_size, hipStream_t stream) {
  (void)in_sizes; (void)n_in; (void)out_size; (void)ws_size;

  // dict order: 0:x_t 1:h0 2:c0 3:We 4:be 5:Wf 6:bf 7:Wi 8:bi 9:Wg 10:bg
  //             11:Wo 12:bo 13:Rf 14:Ri 15:Rg 16:Ro
  char* p = (char*)d_ws;
  size_t off = 0;
  auto take = [&](size_t bytes) -> void* {
    void* r = p + off;
    off = (off + bytes + 255) & ~(size_t)255;
    return r;
  };
  bf16_t* xbf   = (bf16_t*)take((size_t)kM * kDIM * 2);   //  32 MB
  bf16_t* Webf  = (bf16_t*)take((size_t)kE * kDIM * 2);
  bf16_t* Wcat  = (bf16_t*)take((size_t)kNH * kE * 2);    // [4H,E]
  bf16_t* Rcat  = (bf16_t*)take((size_t)kNH * kH * 2);    // [4H,H]
  float*  bcat  = (float*) take((size_t)kNH * 4);
  bf16_t* xebf  = (bf16_t*)take((size_t)kM * kE * 2);     //  64 MB
  float*  xsbuf = (float*) take((size_t)kM * kNH * 4);    // 512 MB
  bf16_t* hb0   = (bf16_t*)take((size_t)kB * kH * 2);
  bf16_t* hb1   = (bf16_t*)take((size_t)kB * kH * 2);
  float*  cst   = (float*) take((size_t)kB * kH * 4);

  auto cvt = [&](const void* src, bf16_t* dst, int n) {
    f32_to_bf16_kernel<<<(n + 255) / 256, 256, 0, stream>>>((const float*)src,
                                                            dst, n);
  };
  auto cpy = [&](const void* src, float* dst, int n) {
    copy_f32_kernel<<<(n + 255) / 256, 256, 0, stream>>>((const float*)src,
                                                         dst, n);
  };

  // Pack weights (gate order i,f,g,o) + activations to bf16.
  cvt(d_in[0],  xbf, kM * kDIM);
  cvt(d_in[3],  Webf, kE * kDIM);
  cvt(d_in[7],  Wcat + 0 * (size_t)kH * kE, kH * kE);   // Wi
  cvt(d_in[5],  Wcat + 1 * (size_t)kH * kE, kH * kE);   // Wf
  cvt(d_in[9],  Wcat + 2 * (size_t)kH * kE, kH * kE);   // Wg
  cvt(d_in[11], Wcat + 3 * (size_t)kH * kE, kH * kE);   // Wo
  cvt(d_in[14], Rcat + 0 * (size_t)kH * kH, kH * kH);   // Ri
  cvt(d_in[13], Rcat + 1 * (size_t)kH * kH, kH * kH);   // Rf
  cvt(d_in[15], Rcat + 2 * (size_t)kH * kH, kH * kH);   // Rg
  cvt(d_in[16], Rcat + 3 * (size_t)kH * kH, kH * kH);   // Ro
  cvt(d_in[1],  hb0, kB * kH);                          // h0 -> bf16
  cpy(d_in[8],  bcat + 0 * kH, kH);                     // bi
  cpy(d_in[6],  bcat + 1 * kH, kH);                     // bf
  cpy(d_in[10], bcat + 2 * kH, kH);                     // bg
  cpy(d_in[12], bcat + 3 * kH, kH);                     // bo
  cpy(d_in[2],  cst, kB * kH);                          // c0

  // Embedding GEMM: xe = x @ We^T + be   (bf16 out)
  gemm_bf16_kernel<true><<<kM / 16, 256, 16 * kDIM * 2, stream>>>(
      xbf, Webf, (const float*)d_in[4], xebf, kE, kDIM);

  // Fused input projections: xs = xe @ [Wi;Wf;Wg;Wo]^T + bcat  (f32 out)
  gemm_bf16_kernel<false><<<kM / 16, 256, 16 * kE * 2, stream>>>(
      xebf, Wcat, bcat, xsbuf, kNH, kE);

  // Sequential recurrence: 512 step launches (graph replay amortizes).
  float* hseq = (float*)d_out;
  for (int t = 0; t < kT; ++t) {
    const bf16_t* hp = (t & 1) ? hb1 : hb0;
    bf16_t*       hn = (t & 1) ? hb0 : hb1;
    lstm_step_kernel<<<dim3(kB / 16, kH / 32), 128, 0, stream>>>(
        hp, hn, Rcat, xsbuf, cst, hseq, t);
  }

  float* hlast = hseq + (size_t)kB * kT * kH;
  float* clast = hlast + (size_t)kB * kH;
  finalize_kernel<<<(kB * kH + 255) / 256, 256, 0, stream>>>(hseq, cst, hlast,
                                                             clast);
}
